// STDiTBlock_81063212745640
// MI455X (gfx1250) — compile-verified, matrix operand order checked by reading, not run on hardware
//
#include <hip/hip_runtime.h>
#include <hip/hip_bf16.h>
#include <math.h>

// ---------------------------------------------------------------------------
// STDiT block for MI455X (gfx1250, wave32, WMMA).
// All GEMM operands are materialized once as f16 (LN / epilogues / one-time
// weight casts), so the WMMA GEMM inner loop is pure b128 copies + fragment
// ds-loads + v_wmma_f32_16x16x32_f16, double-buffered in LDS with a single
// barrier per K-step. Temporal attention (T=16) is a fused one-wave-per-head
// kernel: WMMA scores -> shuffle softmax -> LDS re-fragment -> WMMA scores@V.
// ---------------------------------------------------------------------------

typedef __attribute__((ext_vector_type(16))) _Float16 v16h;
typedef __attribute__((ext_vector_type(8)))  float    v8f;

union AF { v16h v; unsigned int u[8]; _Float16 h[16]; };

#define CDIM   512
#define ADA_ST 3072   // 6*C

__device__ __forceinline__ float gelu_tanh(float x) {
    return 0.5f * x * (1.f + tanhf(0.7978845608028654f * (x + 0.044715f * x * x * x)));
}

__device__ __forceinline__ float block_reduce(float v, float* sm, int domax) {
    int tid = threadIdx.x;
    sm[tid] = v; __syncthreads();
    for (int s = 128; s > 0; s >>= 1) {
        if (tid < s) sm[tid] = domax ? fmaxf(sm[tid], sm[tid + s]) : (sm[tid] + sm[tid + s]);
        __syncthreads();
    }
    float r = sm[0]; __syncthreads();
    return r;
}

// K index pattern for the 16-bit A/B fragment (16x32), per CDNA5 ISA 7.12.2.
__device__ __forceinline__ int frag_kbase(int vv, int hl) {
    return (vv < 4 ? 2 * vv : 16 + 2 * (vv - 4)) + 8 * hl;
}

// ---------------------------------------------------------------------------
// Generic batched f16 GEMM:  C[z] = epilogue( scale * A[z] @ B[z] + bias )
//   A: (M,K) f16 row-major. B: (K,N) f16 (bT=0) or (N,K) f16 (bT=1).
//   batch z = zb*nh + zh with independent strides (head-sliced views).
//   epi: 0 bias, 1 bias+GELU, 2 res + gate*(..) (f32 out).  outHalf: f16 C.
//   Double-buffered LDS, one barrier per K-step.
// ---------------------------------------------------------------------------
__global__ __launch_bounds__(256) void gemm_wmma(
    const _Float16* __restrict__ A, long long sAb, long long sAh, int lda,
    const _Float16* __restrict__ B, long long sBb, long long sBh, int ldb, int bT,
    void* __restrict__ Cv, long long sCb, long long sCh, int ldc, int outHalf,
    const float* __restrict__ bias,
    int M, int N, int K, int nh,
    int epi, const float* __restrict__ res,
    const float* __restrict__ gate, int rpg, float scale)
{
    __shared__ _Float16 lA[2][128][40];
    __shared__ _Float16 lB[2][128][40];

    int z  = blockIdx.z;
    int zb = z / nh, zh = z % nh;
    A += zb * sAb + zh * sAh;
    B += zb * sBb + zh * sBh;
    long long coff = zb * sCb + zh * sCh;

    int tid  = threadIdx.x;
    int m0   = blockIdx.y * 128;
    int n0   = blockIdx.x * 128;
    int wid  = tid >> 5;
    int lane = tid & 31;
    int hl   = lane >> 4;
    int lm   = lane & 15;
    int wm   = (wid & 3) * 32;   // 2 m-tiles
    int wn   = (wid >> 2) * 64;  // 4 n-tiles

    v8f acc[2][4] = {};
    uint4 ra[2], rb[2];

    auto gloadA = [&](int k0, uint4* r) {
        int rr = tid >> 1, kh = (tid & 1) * 16;
        int gr = m0 + rr;
        if (gr < M) {
            const uint4* p = reinterpret_cast<const uint4*>(A + (long long)gr * lda + k0 + kh);
            r[0] = p[0]; r[1] = p[1];
        } else {
            r[0] = uint4{0, 0, 0, 0}; r[1] = uint4{0, 0, 0, 0};
        }
    };
    auto gloadB = [&](int k0, uint4* r) {
        if (bT) {
            int rr = tid >> 1, kh = (tid & 1) * 16;
            int gn = n0 + rr;
            if (gn < N) {
                const uint4* p = reinterpret_cast<const uint4*>(B + (long long)gn * ldb + k0 + kh);
                r[0] = p[0]; r[1] = p[1];
            } else { r[0] = uint4{0, 0, 0, 0}; r[1] = uint4{0, 0, 0, 0}; }
        } else {
            int kk = tid >> 3, nb = (tid & 7) * 16;
            int gn = n0 + nb;
            if (gn < N) {
                const uint4* p = reinterpret_cast<const uint4*>(B + (long long)(k0 + kk) * ldb + gn);
                r[0] = p[0]; r[1] = p[1];
            } else { r[0] = uint4{0, 0, 0, 0}; r[1] = uint4{0, 0, 0, 0}; }
        }
    };
    auto storeTiles = [&](int buf) {
        int rr = tid >> 1, kh = (tid & 1) * 16;
        *reinterpret_cast<uint4*>(&lA[buf][rr][kh])     = ra[0];
        *reinterpret_cast<uint4*>(&lA[buf][rr][kh + 8]) = ra[1];
        if (bT) {
            *reinterpret_cast<uint4*>(&lB[buf][rr][kh])     = rb[0];
            *reinterpret_cast<uint4*>(&lB[buf][rr][kh + 8]) = rb[1];
        } else {
            int kk = tid >> 3, nb = (tid & 7) * 16;
            const _Float16* h = reinterpret_cast<const _Float16*>(rb);
            #pragma unroll
            for (int i = 0; i < 16; ++i) lB[buf][nb + i][kk] = h[i];
        }
    };

    int nk = K >> 5;
    gloadA(0, ra); gloadB(0, rb);

    for (int it = 0; it < nk; ++it) {
        int buf = it & 1;
        storeTiles(buf);
        __syncthreads();
        if (it + 1 < nk) { gloadA((it + 1) << 5, ra); gloadB((it + 1) << 5, rb); }

        AF af[2], bf[4];
        #pragma unroll
        for (int mt = 0; mt < 2; ++mt) {
            int m = wm + mt * 16 + lm;
            #pragma unroll
            for (int vv = 0; vv < 8; ++vv)
                af[mt].u[vv] = *(const unsigned int*)&lA[buf][m][frag_kbase(vv, hl)];
        }
        #pragma unroll
        for (int nt = 0; nt < 4; ++nt) {
            int n = wn + nt * 16 + lm;
            #pragma unroll
            for (int vv = 0; vv < 8; ++vv)
                bf[nt].u[vv] = *(const unsigned int*)&lB[buf][n][frag_kbase(vv, hl)];
        }
        #pragma unroll
        for (int mt = 0; mt < 2; ++mt)
            #pragma unroll
            for (int nt = 0; nt < 4; ++nt)
                acc[mt][nt] = __builtin_amdgcn_wmma_f32_16x16x32_f16(
                    false, af[mt].v, false, bf[nt].v, (short)0, acc[mt][nt], false, false);
    }

    // ---- epilogue ----
    float*     Cf = (float*)Cv + coff;
    _Float16*  Ch = (_Float16*)Cv + coff;
    #pragma unroll
    for (int mt = 0; mt < 2; ++mt) {
        #pragma unroll
        for (int nt = 0; nt < 4; ++nt) {
            int col = n0 + wn + nt * 16 + lm;
            if (col >= N) continue;
            float bv = bias ? bias[col] : 0.f;
            #pragma unroll
            for (int r = 0; r < 8; ++r) {
                int row = m0 + wm + mt * 16 + hl * 8 + r;
                if (row >= M) continue;
                float v = acc[mt][nt][r] * scale + bv;
                if (epi == 1) {
                    v = gelu_tanh(v);
                } else if (epi == 2) {
                    float g = gate[(row / rpg) * ADA_ST + col];
                    v = res[(long long)row * ldc + col] + g * v;
                }
                if (outHalf) Ch[(long long)row * ldc + col] = (_Float16)v;
                else         Cf[(long long)row * ldc + col] = v;
            }
        }
    }
}

// ---------------------------------------------------------------------------
// Fused temporal diff-attention. One wave per (b*S, head) batch, T = 16.
// Operands are f16 -> packed-half u32 loads feed fragments directly.
// Output written as O[bt*65536 + t*4096 + h*512 + d]  (rows (bt,t), cols h*d)
// ---------------------------------------------------------------------------
__global__ __launch_bounds__(256) void temporal_attn(
    const _Float16* __restrict__ Q, const _Float16* __restrict__ Kb,
    const _Float16* __restrict__ V, float* __restrict__ O,
    const float* __restrict__ lamp)
{
    __shared__ _Float16 lS[8][16][34];

    int w    = threadIdx.x >> 5;
    int z    = blockIdx.x * 8 + w;
    int lane = threadIdx.x & 31;
    int hl   = lane >> 4;
    int lm   = lane & 15;

    const _Float16* q  = Q  + (long long)z * 16384;
    const _Float16* kp = Kb + (long long)z * 16384;
    const _Float16* vp = V  + (long long)z * 8192;
    float lam = *lamp;

    v8f s1 = {}, s2 = {};
    for (int kt = 0; kt < 16; ++kt) {
        int k0 = kt * 32;
        AF a1, a2, b1, b2;
        #pragma unroll
        for (int vv = 0; vv < 8; ++vv) {
            int kb = frag_kbase(vv, hl) + k0;
            a1.u[vv] = *(const unsigned int*)(q  + lm * 1024 + kb);
            a2.u[vv] = *(const unsigned int*)(q  + lm * 1024 + 512 + kb);
            b1.u[vv] = *(const unsigned int*)(kp + lm * 1024 + kb);
            b2.u[vv] = *(const unsigned int*)(kp + lm * 1024 + 512 + kb);
        }
        s1 = __builtin_amdgcn_wmma_f32_16x16x32_f16(false, a1.v, false, b1.v, (short)0, s1, false, false);
        s2 = __builtin_amdgcn_wmma_f32_16x16x32_f16(false, a2.v, false, b2.v, (short)0, s2, false, false);
    }

    // softmax over key dim (16 lanes within a half-wave), per accumulator row
    float comb[8];
    #pragma unroll
    for (int r = 0; r < 8; ++r) {
        float x1 = s1[r], x2 = s2[r];
        float m1 = x1, m2 = x2;
        #pragma unroll
        for (int m = 1; m < 16; m <<= 1) {
            m1 = fmaxf(m1, __shfl_xor(m1, m, 32));
            m2 = fmaxf(m2, __shfl_xor(m2, m, 32));
        }
        float e1 = expf(x1 - m1), e2 = expf(x2 - m2);
        float t1 = e1, t2 = e2;
        #pragma unroll
        for (int m = 1; m < 16; m <<= 1) {
            t1 += __shfl_xor(t1, m, 32);
            t2 += __shfl_xor(t2, m, 32);
        }
        comb[r] = e1 / t1 - lam * (e2 / t2);
    }

    // re-fragment scores through LDS (C layout -> A layout, K padded to 32)
    #pragma unroll
    for (int r = 0; r < 8; ++r) lS[w][lm][16 + hl * 8 + r] = (_Float16)0.f;
    #pragma unroll
    for (int r = 0; r < 8; ++r) lS[w][hl * 8 + r][lm] = (_Float16)comb[r];

    AF af;
    #pragma unroll
    for (int vv = 0; vv < 8; ++vv)
        af.u[vv] = *(const unsigned int*)&lS[w][lm][frag_kbase(vv, hl)];

    int bt = z >> 3, h = z & 7;
    float* ob = O + (long long)bt * 65536 + (long long)h * 512;
    for (int nt = 0; nt < 32; ++nt) {
        int d0 = nt * 16;
        AF bfv;
        #pragma unroll
        for (int vv = 0; vv < 8; ++vv) {
            int kb = frag_kbase(vv, hl);
            bfv.h[2 * vv]     = (kb     < 16) ? vp[(kb    ) * 512 + d0 + lm] : (_Float16)0.f;
            bfv.h[2 * vv + 1] = (kb + 1 < 16) ? vp[(kb + 1) * 512 + d0 + lm] : (_Float16)0.f;
        }
        v8f zz = {};
        v8f o = __builtin_amdgcn_wmma_f32_16x16x32_f16(false, af.v, false, bfv.v, (short)0, zz, false, false);
        #pragma unroll
        for (int r = 0; r < 8; ++r)
            ob[(long long)(hl * 8 + r) * 4096 + d0 + lm] = o[r];
    }
}

// ---------------------------------------------------------------------------
// one-time f32 -> f16 cast (n multiple of 4)
// ---------------------------------------------------------------------------
__global__ __launch_bounds__(256) void cast_half(
    const float* __restrict__ in, _Float16* __restrict__ out, long long n)
{
    long long i = ((long long)blockIdx.x * 256 + threadIdx.x) * 4;
    if (i >= n) return;
    float4 f = *(const float4*)(in + i);
    union { _Float16 h[4]; uint2 u; } p;
    p.h[0] = (_Float16)f.x; p.h[1] = (_Float16)f.y;
    p.h[2] = (_Float16)f.z; p.h[3] = (_Float16)f.w;
    *(uint2*)(out + i) = p.u;
}

// ---------------------------------------------------------------------------
// adaLN: out = silu(c) @ W + b     (2 x 3072, f32)
// ---------------------------------------------------------------------------
__global__ __launch_bounds__(256) void ada_kernel(
    const float* __restrict__ c, const float* __restrict__ W,
    const float* __restrict__ b, float* __restrict__ out)
{
    int j = blockIdx.x * 256 + threadIdx.x;
    if (j >= 2 * ADA_ST) return;
    int bb = j / ADA_ST, col = j % ADA_ST;
    float acc = b[col];
    for (int k = 0; k < CDIM; ++k) {
        float x = c[bb * CDIM + k];
        x = x / (1.f + expf(-x));
        acc += x * W[(long long)k * ADA_ST + col];
    }
    out[j] = acc;
}

// ---------------------------------------------------------------------------
// LayerNorm over C=512, optional pre/post modulation, writes f16 (GEMM A).
// ---------------------------------------------------------------------------
__global__ __launch_bounds__(256) void ln_mod(
    const float* __restrict__ in, _Float16* __restrict__ out,
    const float* __restrict__ ada, int rowsPerB,
    int preShift, int preScale, int postShift, int postScale)
{
    __shared__ float sm[256];
    int row = blockIdx.x, tid = threadIdx.x;
    const float* ab = ada ? ada + (long long)(row / rowsPerB) * ADA_ST : nullptr;
    float v0 = in[(long long)row * CDIM + tid];
    float v1 = in[(long long)row * CDIM + tid + 256];
    if (preScale >= 0) {
        v0 = v0 * (1.f + ab[preScale + tid])       + ab[preShift + tid];
        v1 = v1 * (1.f + ab[preScale + tid + 256]) + ab[preShift + tid + 256];
    }
    float mu  = block_reduce(v0 + v1, sm, 0) * (1.f / 512.f);
    float d0 = v0 - mu, d1 = v1 - mu;
    float var = block_reduce(d0 * d0 + d1 * d1, sm, 0) * (1.f / 512.f);
    float rs  = rsqrtf(var + 1e-6f);
    v0 = d0 * rs; v1 = d1 * rs;
    if (postScale >= 0) {
        v0 = v0 * (1.f + ab[postScale + tid])       + ab[postShift + tid];
        v1 = v1 * (1.f + ab[postScale + tid + 256]) + ab[postShift + tid + 256];
    }
    out[(long long)row * CDIM + tid]       = (_Float16)v0;
    out[(long long)row * CDIM + tid + 256] = (_Float16)v1;
}

// ---------------------------------------------------------------------------
// out(f16) = softmax(S1) [- lam * softmax(S2)], row-wise, len in {256,512}
// ---------------------------------------------------------------------------
__global__ __launch_bounds__(256) void softmax_combine(
    const float* __restrict__ S1, const float* __restrict__ S2,
    _Float16* __restrict__ Out, const float* __restrict__ lamp, int len)
{
    __shared__ float sm[256];
    int row = blockIdx.x, tid = threadIdx.x;
    int n = len >> 8;
    const float* r1 = S1 + (long long)row * len;
    float a[2], e1[2];
    float mx = -3.4e38f;
    for (int i = 0; i < n; ++i) { a[i] = r1[tid + (i << 8)]; mx = fmaxf(mx, a[i]); }
    mx = block_reduce(mx, sm, 1);
    float s = 0.f;
    for (int i = 0; i < n; ++i) { e1[i] = expf(a[i] - mx); s += e1[i]; }
    s = block_reduce(s, sm, 0);
    float inv1 = 1.f / s;

    if (S2) {
        const float* r2 = S2 + (long long)row * len;
        float b[2], e2[2];
        float mx2 = -3.4e38f;
        for (int i = 0; i < n; ++i) { b[i] = r2[tid + (i << 8)]; mx2 = fmaxf(mx2, b[i]); }
        mx2 = block_reduce(mx2, sm, 1);
        float s2 = 0.f;
        for (int i = 0; i < n; ++i) { e2[i] = expf(b[i] - mx2); s2 += e2[i]; }
        s2 = block_reduce(s2, sm, 0);
        float inv2 = 1.f / s2, lam = *lamp;
        for (int i = 0; i < n; ++i)
            Out[(long long)row * len + tid + (i << 8)] = (_Float16)(e1[i] * inv1 - lam * (e2[i] * inv2));
    } else {
        for (int i = 0; i < n; ++i)
            Out[(long long)row * len + tid + (i << 8)] = (_Float16)(e1[i] * inv1);
    }
}

// ---------------------------------------------------------------------------
// Group-norm over (nI x 512) per group (group = outer*8 + h), generic strides.
// gn_apply writes the normalized result as f16 (next GEMM's A operand).
// ---------------------------------------------------------------------------
__global__ __launch_bounds__(256) void gn_stats(
    const float* __restrict__ O, float* __restrict__ st,
    long long sOuter, long long sH, long long sI, int nI)
{
    __shared__ float sm[256];
    int g = blockIdx.x;
    int outer = g >> 3, h = g & 7;
    const float* base = O + (long long)outer * sOuter + (long long)h * sH;
    int tot = nI * 512;
    float s = 0.f, sq = 0.f;
    for (int i = threadIdx.x; i < tot; i += 256) {
        float v = base[(long long)(i >> 9) * sI + (i & 511)];
        s += v; sq += v * v;
    }
    s  = block_reduce(s, sm, 0);
    sq = block_reduce(sq, sm, 0);
    if (threadIdx.x == 0) {
        float mu = s / tot;
        st[2 * g]     = mu;
        st[2 * g + 1] = rsqrtf(sq / tot - mu * mu + 1e-5f);
    }
}

__global__ __launch_bounds__(256) void gn_apply(
    const float* __restrict__ O, _Float16* __restrict__ OH,
    const float* __restrict__ st,
    const float* __restrict__ gnw, const float* __restrict__ gnb,
    const float* __restrict__ lamp,
    long long sOuter, long long sH, long long sI, int nI, int ngroups)
{
    long long idx = (long long)blockIdx.x * 256 + threadIdx.x;
    long long per = (long long)nI * 512;
    if (idx >= (long long)ngroups * per) return;
    int g = (int)(idx / per);
    long long rem = idx % per;
    int i = (int)(rem >> 9), d = (int)(rem & 511);
    int outer = g >> 3, h = g & 7;
    long long a = (long long)outer * sOuter + (long long)h * sH + (long long)i * sI + d;
    float lam = *lamp;
    OH[a] = (_Float16)(((O[a] - st[2 * g]) * st[2 * g + 1] * gnw[h] + gnb[h]) * (1.f - lam));
}

// ---------------------------------------------------------------------------
// (b,T,S,C) -> (b,S,T,C) copy, and the inverse-permute gated residual add.
// ---------------------------------------------------------------------------
__global__ __launch_bounds__(256) void transpose_bts(
    const float* __restrict__ in, float* __restrict__ out)
{
    int row = blockIdx.x;                 // out row in (b,s,t) order
    int b = row >> 12, s = (row >> 4) & 255, t = row & 15;
    const float* ir = in + (long long)((b << 12) + t * 256 + s) * CDIM;
    float* orow = out + (long long)row * CDIM;
    for (int c = threadIdx.x; c < CDIM; c += 256) orow[c] = ir[c];
}

__global__ __launch_bounds__(256) void permute_gated_add(
    float* __restrict__ X, const float* __restrict__ Tt,
    const float* __restrict__ gate)
{
    int row = blockIdx.x;                 // X row in (b,t,s) order
    int b = row >> 12, t = (row >> 8) & 15, s = row & 255;
    const float* ir = Tt + (long long)((b << 12) + s * 16 + t) * CDIM;
    float* xr = X + (long long)row * CDIM;
    for (int c = threadIdx.x; c < CDIM; c += 256)
        xr[c] += gate[b * ADA_ST + c] * ir[c];
}

// ---------------------------------------------------------------------------
// Host orchestration
// ---------------------------------------------------------------------------
static void launch_gemm(hipStream_t st,
    const _Float16* A, long long sAb, long long sAh, int lda,
    const _Float16* B, long long sBb, long long sBh, int ldb, int bT,
    void* C, long long sCb, long long sCh, int ldc, int outHalf,
    const float* bias, int M, int N, int K, int nb, int nh,
    int epi, const float* res, const float* gate, int rpg, float scale)
{
    dim3 g((N + 127) / 128, (M + 127) / 128, nb * nh);
    gemm_wmma<<<g, 256, 0, st>>>(A, sAb, sAh, lda, B, sBb, sBh, ldb, bT,
                                 C, sCb, sCh, ldc, outHalf, bias, M, N, K, nh,
                                 epi, res, gate, rpg, scale);
}

static void launch_cast(hipStream_t st, const float* in, _Float16* out, long long n)
{
    cast_half<<<(unsigned)((n / 4 + 255) / 256), 256, 0, st>>>(in, out, n);
}

extern "C" void kernel_launch(void* const* d_in, const int* in_sizes, int n_in,
                              void* d_out, int out_size, void* d_ws, size_t ws_size,
                              hipStream_t stream)
{
    (void)in_sizes; (void)n_in; (void)out_size; (void)ws_size;
    const float* x_in   = (const float*)d_in[0];
    const float* c_in   = (const float*)d_in[1];
    const float* steps  = (const float*)d_in[2];
    const float* lam_s  = (const float*)d_in[3];
    const float* lam_t  = (const float*)d_in[4];
    const float* ada_w  = (const float*)d_in[5];
    const float* ada_b  = (const float*)d_in[6];
    const float* s_wq = (const float*)d_in[7],  *s_bq = (const float*)d_in[8];
    const float* s_wk = (const float*)d_in[9],  *s_bk = (const float*)d_in[10];
    const float* s_wv = (const float*)d_in[11], *s_bv = (const float*)d_in[12];
    const float* s_wo = (const float*)d_in[13], *s_bo = (const float*)d_in[14];
    const float* s_gnw = (const float*)d_in[15], *s_gnb = (const float*)d_in[16];
    const float* t_wq = (const float*)d_in[17], *t_bq = (const float*)d_in[18];
    const float* t_wk = (const float*)d_in[19], *t_bk = (const float*)d_in[20];
    const float* t_wv = (const float*)d_in[21], *t_bv = (const float*)d_in[22];
    const float* t_wo = (const float*)d_in[23], *t_bo = (const float*)d_in[24];
    const float* t_gnw = (const float*)d_in[25], *t_gnb = (const float*)d_in[26];
    const float* c_wq  = (const float*)d_in[27], *c_bq  = (const float*)d_in[28];
    const float* c_wkv = (const float*)d_in[29], *c_bkv = (const float*)d_in[30];
    const float* c_wp  = (const float*)d_in[31], *c_bp  = (const float*)d_in[32];
    const float* m_w1 = (const float*)d_in[33], *m_b1 = (const float*)d_in[34];
    const float* m_w2 = (const float*)d_in[35], *m_b2 = (const float*)d_in[36];
    float* out = (float*)d_out;

    // ---- workspace layout ----
    float* W = (float*)d_ws;
    size_t off = 0;
    auto takeF = [&](size_t n) { float* p = W + off; off += n; return p; };
    float* XB  = takeF(8192ull * 512);     // running x (f32)
    float* TT  = takeF(8192ull * 512);     // (b,S,T,C) temp (f32)
    float* ADA = takeF(8192);              // 2 x 3072
    float* S1  = takeF(65536ull * 256);    // f32 logits (cross spans S1+S2)
    float* S2  = takeF(65536ull * 256);
    float* O   = takeF(8192ull * 4096);    // f32 attention output (pre-GN)
    float* ST  = takeF(8192);              // group-norm stats
    (void)S2;

    _Float16* Hb = (_Float16*)(W + off);
    size_t hoff = 0;
    auto takeH = [&](size_t n) { _Float16* p = Hb + hoff; hoff += n; return p; };
    _Float16* QH   = takeH(8192ull * 8192);  // q proj / cross q
    _Float16* KH   = takeH(8192ull * 8192);  // k proj / cross kv
    _Float16* VH   = takeH(8192ull * 4096);  // v proj
    _Float16* OH   = takeH(8192ull * 4096);  // post-GN attn out / cross o
    _Float16* SH   = takeH(65536ull * 512);  // softmax probs / MLP hidden
    _Float16* LNh  = takeH(8192ull * 512);   // LN output (GEMM A)
    _Float16* CLNh = takeH(1024ull * 512);   // LN(initial_steps)
    _Float16* WqH  = takeH(512ull * 8192);   // weights (re-cast per phase)
    _Float16* WkH  = takeH(512ull * 8192);
    _Float16* WvH  = takeH(512ull * 4096);
    _Float16* WoH  = takeH(4096ull * 512);
    _Float16* WcqH  = takeH(512ull * 512);
    _Float16* WckvH = takeH(512ull * 1024);
    _Float16* WcpH  = takeH(512ull * 512);
    _Float16* Wm1H  = takeH(512ull * 2048);
    _Float16* Wm2H  = takeH(2048ull * 512);

    // 1) adaLN + one-time small-weight casts
    ada_kernel<<<24, 256, 0, stream>>>(c_in, ada_w, ada_b, ADA);
    launch_cast(stream, c_wq,  WcqH,  512ll * 512);
    launch_cast(stream, c_wkv, WckvH, 512ll * 1024);
    launch_cast(stream, c_wp,  WcpH,  512ll * 512);
    launch_cast(stream, m_w1,  Wm1H,  512ll * 2048);
    launch_cast(stream, m_w2,  Wm2H,  2048ll * 512);

    // ---- spatial diff attention ----
    launch_cast(stream, s_wq, WqH, 512ll * 8192);
    launch_cast(stream, s_wk, WkH, 512ll * 8192);
    launch_cast(stream, s_wv, WvH, 512ll * 4096);
    launch_cast(stream, s_wo, WoH, 4096ll * 512);
    ln_mod<<<8192, 256, 0, stream>>>(x_in, LNh, ADA, 4096, 0, 512, -1, -1);
    launch_gemm(stream, LNh,0,0,512, WqH,0,0,8192,0, QH,0,0,8192,1, s_bq, 8192,8192,512, 1,1, 0,nullptr,nullptr,1, 1.f);
    launch_gemm(stream, LNh,0,0,512, WkH,0,0,8192,0, KH,0,0,8192,1, s_bk, 8192,8192,512, 1,1, 0,nullptr,nullptr,1, 1.f);
    launch_gemm(stream, LNh,0,0,512, WvH,0,0,4096,0, VH,0,0,4096,1, s_bv, 8192,4096,512, 1,1, 0,nullptr,nullptr,1, 1.f);
    // scores1 = q1 @ k1^T, scores2 = q2 @ k2^T   (batched over 256 (b*T,h))
    launch_gemm(stream, QH,      262144,0,1024, KH,      262144,0,1024,1, S1,65536,0,256,0, nullptr, 256,256,512, 256,1, 0,nullptr,nullptr,1, 1.f);
    launch_gemm(stream, QH + 512,262144,0,1024, KH + 512,262144,0,1024,1, S2,65536,0,256,0, nullptr, 256,256,512, 256,1, 0,nullptr,nullptr,1, 1.f);
    softmax_combine<<<65536, 256, 0, stream>>>(S1, S2, SH, lam_s, 256);
    launch_gemm(stream, SH,65536,0,256, VH,131072,0,512,0, O,131072,0,512,0, nullptr, 256,512,256, 256,1, 0,nullptr,nullptr,1, 1.f);
    gn_stats<<<256, 256, 0, stream>>>(O, ST, 1048576LL, 131072LL, 512LL, 256);
    gn_apply<<<131072, 256, 0, stream>>>(O, OH, ST, s_gnw, s_gnb, lam_s, 1048576LL, 131072LL, 512LL, 256, 256);
    // wo + gated residual:  XB = x + gate_msa * (o @ wo + bo)
    launch_gemm(stream, OH,0,0,4096, WoH,0,0,512,0, XB,0,0,512,0, s_bo, 8192,512,4096, 1,1, 2, x_in, ADA + 1024, 4096, 1.f);

    // ---- temporal diff attention ----
    launch_cast(stream, t_wq, WqH, 512ll * 8192);
    launch_cast(stream, t_wk, WkH, 512ll * 8192);
    launch_cast(stream, t_wv, WvH, 512ll * 4096);
    launch_cast(stream, t_wo, WoH, 4096ll * 512);
    transpose_bts<<<8192, 256, 0, stream>>>(XB, TT);
    ln_mod<<<8192, 256, 0, stream>>>(TT, LNh, nullptr, 4096, -1, -1, -1, -1);
    launch_gemm(stream, LNh,0,0,512, WqH,0,0,8192,0, QH,0,0,8192,1, t_bq, 8192,8192,512, 1,1, 0,nullptr,nullptr,1, 1.f);
    launch_gemm(stream, LNh,0,0,512, WkH,0,0,8192,0, KH,0,0,8192,1, t_bk, 8192,8192,512, 1,1, 0,nullptr,nullptr,1, 1.f);
    launch_gemm(stream, LNh,0,0,512, WvH,0,0,4096,0, VH,0,0,4096,1, t_bv, 8192,4096,512, 1,1, 0,nullptr,nullptr,1, 1.f);
    temporal_attn<<<512, 256, 0, stream>>>(QH, KH, VH, O, lam_t);
    gn_stats<<<4096, 256, 0, stream>>>(O, ST, 65536LL, 512LL, 4096LL, 16);
    gn_apply<<<131072, 256, 0, stream>>>(O, OH, ST, t_gnw, t_gnb, lam_t, 65536LL, 512LL, 4096LL, 16, 4096);
    launch_gemm(stream, OH,0,0,4096, WoH,0,0,512,0, TT,0,0,512,0, t_bo, 8192,512,4096, 1,1, 0,nullptr,nullptr,1, 1.f);
    permute_gated_add<<<8192, 256, 0, stream>>>(XB, TT, ADA + 1024);

    // ---- cross attention ----
    ln_mod<<<8192, 256, 0, stream>>>(XB, LNh, nullptr, 4096, -1, -1, -1, -1);
    ln_mod<<<1024, 256, 0, stream>>>(steps, CLNh, nullptr, 512, -1, -1, -1, -1);
    launch_gemm(stream, LNh,0,0,512, WcqH,0,0,512,0, QH,0,0,512,1, c_bq, 8192,512,512, 1,1, 0,nullptr,nullptr,1, 1.f);
    launch_gemm(stream, CLNh,0,0,512, WckvH,0,0,1024,0, KH,0,0,1024,1, c_bkv, 1024,1024,512, 1,1, 0,nullptr,nullptr,1, 1.f);
    // scores (b=2, h=8): q (4096x64 slices), k rows from kv; scale = d^-1/2
    launch_gemm(stream, QH, 2097152,64,512, KH, 524288,64,1024,1, S1, 16777216,2097152,512,0, nullptr,
                4096,512,64, 2,8, 0,nullptr,nullptr,1, 0.125f);
    softmax_combine<<<65536, 256, 0, stream>>>(S1, nullptr, SH, lam_s, 512);
    launch_gemm(stream, SH,16777216,2097152,512, KH + 512,524288,64,1024,0, OH,2097152,64,512,1, nullptr,
                4096,64,512, 2,8, 0,nullptr,nullptr,1, 1.f);
    launch_gemm(stream, OH,0,0,512, WcpH,0,0,512,0, XB,0,0,512,0, c_bp, 8192,512,512, 1,1, 2, XB, ADA + 1024, 4096, 1.f);

    // ---- MLP ----
    ln_mod<<<8192, 256, 0, stream>>>(XB, LNh, ADA, 4096, -1, -1, 1536, 2048);
    launch_gemm(stream, LNh,0,0,512, Wm1H,0,0,2048,0, SH,0,0,2048,1, m_b1, 8192,2048,512, 1,1, 1, nullptr,nullptr,1, 1.f);
    launch_gemm(stream, SH,0,0,2048, Wm2H,0,0,512,0, out,0,0,512,0, m_b2, 8192,512,2048, 1,1, 2, XB, ADA + 2560, 4096, 1.f);
}
